// CompositionalTwoArmedAgent_9431748182598
// MI455X (gfx1250) — matrix operand order repaired and checked by reading, not compiled, not run on hardware
//
#include <hip/hip_runtime.h>
#include <hip/hip_bf16.h>
#include <cstdint>
#include <math.h>

typedef __attribute__((ext_vector_type(16))) _Float16 v16h;
typedef __attribute__((ext_vector_type(8)))  float    v8f;
typedef int v4i __attribute__((vector_size(16)));

#define GLOBAL_AS __attribute__((address_space(1)))
#define LDS_AS    __attribute__((address_space(3)))

#if defined(__has_builtin)
#  if __has_builtin(__builtin_amdgcn_global_load_async_to_lds_b128) && \
      __has_builtin(__builtin_amdgcn_global_load_async_to_lds_b32)  && \
      __has_builtin(__builtin_amdgcn_s_wait_asynccnt)
#    define HAS_ASYNC_LDS 1
#  endif
#endif
#ifndef HAS_ASYNC_LDS
#  define HAS_ASYNC_LDS 0
#endif

// Async global->LDS staging helpers (CDNA5 GLOBAL_LOAD_ASYNC_TO_LDS_*; tracked
// by ASYNCcnt). Param types per hipcc diagnostics: (v4i AS1*, v4i AS3*, imm, imm).
__device__ inline void stage_f4(const float* g, float* l)
{
#if HAS_ASYNC_LDS
    __builtin_amdgcn_global_load_async_to_lds_b128(
        (GLOBAL_AS v4i*)g, (LDS_AS v4i*)l, 0, 0);
#else
    *(float4*)l = *(const float4*)g;
#endif
}
__device__ inline void stage_f1(const float* g, float* l)
{
#if HAS_ASYNC_LDS
    __builtin_amdgcn_global_load_async_to_lds_b32(
        (GLOBAL_AS int*)g, (LDS_AS int*)l, 0, 0);
#else
    *l = *g;
#endif
}
__device__ inline void wait_async_all()
{
#if HAS_ASYNC_LDS
    __builtin_amdgcn_s_wait_asynccnt(0);
#endif
}

// ---------------------------------------------------------------------------
// WMMA GEMV:  y[r] = act( dot(W[r,:], x) + bias0[r] + bias1[r] + dot(Wsmall[r,:], xsmall) )
// W is (outDim, K) row-major f32, x is (K) f32. K % 64 == 0, outDim % 16 == 0,
// K <= 2048. One wave32 per 16-output tile; x staged once per block into LDS.
// A-matrix: only row M=0 nonzero (x), same contiguous k order as B. B-matrix: lane n
// carries W row (n0+n); lanes 0-15 hold K 0-15 of the step, lanes 16-31 K 16-31.
// Result: D row M=0 in acc[0] on lanes 0-15.
// ---------------------------------------------------------------------------
__global__ void gemv_wmma_f16(const float* __restrict__ W,
                              const float* __restrict__ x,
                              const float* __restrict__ bias0,
                              const float* __restrict__ bias1,
                              const float* __restrict__ Wsmall,
                              const float* __restrict__ xsmall,
                              int Ksmall,
                              float* __restrict__ y,
                              int outDim, int K, int doRelu)
{
    __shared__ float sx[2048];

    const int tid  = threadIdx.x;
    const int lane = tid & 31;
    const int wave = (blockIdx.x * blockDim.x + tid) >> 5;
    const int n0   = wave * 16;
    if (n0 >= outDim) return;              // wave-uniform (grid sized exactly)

    // Stage x into LDS (async path when available).
    for (int i = tid * 4; i < K; i += blockDim.x * 4)
        stage_f4(x + i, sx + i);
    wait_async_all();
    __syncthreads();

    const int halfSel = lane >> 4;         // 0 = lanes 0-15, 1 = lanes 16-31
    const int nloc    = lane & 15;
    const float* wrow = W + (size_t)(n0 + nloc) * (size_t)K;

    v8f acc = {};
    for (int k0 = 0; k0 < K; k0 += 64) {
        // ---- issue all weight loads for two k-steps (8 x b128 per lane) ----
        float4 f0[4], f1[4];
        {
            const float4* wp0 = (const float4*)(wrow + k0 + halfSel * 16);
            const float4* wp1 = (const float4*)(wrow + k0 + 32 + halfSel * 16);
            #pragma unroll
            for (int i = 0; i < 4; ++i) f0[i] = wp0[i];
            #pragma unroll
            for (int i = 0; i < 4; ++i) f1[i] = wp1[i];
        }
        // ---- A operands from LDS (only matrix row M=0 carries x) ----
        v16h a0 = {}, a1 = {};
        if (nloc == 0) {
            const int xo = k0 + halfSel * 16;
            #pragma unroll
            for (int i = 0; i < 8; ++i) a0[i]     = (_Float16)sx[xo + i];
            #pragma unroll
            for (int i = 0; i < 8; ++i) a0[8 + i] = (_Float16)sx[xo + 8 + i];
            #pragma unroll
            for (int i = 0; i < 8; ++i) a1[i]     = (_Float16)sx[xo + 32 + i];
            #pragma unroll
            for (int i = 0; i < 8; ++i) a1[8 + i] = (_Float16)sx[xo + 40 + i];
        }
        // ---- B operands (f32 -> f16) ----
        v16h b0, b1;
        #pragma unroll
        for (int i = 0; i < 4; ++i) {
            b0[4*i+0] = (_Float16)f0[i].x;  b0[4*i+1] = (_Float16)f0[i].y;
            b0[4*i+2] = (_Float16)f0[i].z;  b0[4*i+3] = (_Float16)f0[i].w;
            b1[4*i+0] = (_Float16)f1[i].x;  b1[4*i+1] = (_Float16)f1[i].y;
            b1[4*i+2] = (_Float16)f1[i].z;  b1[4*i+3] = (_Float16)f1[i].w;
        }
        acc = __builtin_amdgcn_wmma_f32_16x16x32_f16(
                  false, a0, false, b0, (short)0, acc, false, false);
        acc = __builtin_amdgcn_wmma_f32_16x16x32_f16(
                  false, a1, false, b1, (short)0, acc, false, false);
    }

    if (halfSel == 0) {                    // D row M=0 on lanes 0-15, VGPR0
        const int r = n0 + nloc;
        float s = acc[0];
        if (bias0)  s += bias0[r];
        if (bias1)  s += bias1[r];
        if (Wsmall) {
            const float* wr = Wsmall + (size_t)r * (size_t)Ksmall;
            for (int k = 0; k < Ksmall; ++k) s += wr[k] * xsmall[k];
        }
        if (doRelu) s = fmaxf(s, 0.f);
        y[r] = s;
    }
}

// ---------------------------------------------------------------------------
// Init workspace accumulators.
// ---------------------------------------------------------------------------
__global__ void init_k(float* __restrict__ macc, float* __restrict__ scal, int H)
{
    int i = blockIdx.x * blockDim.x + threadIdx.x;
    if (i < H)  macc[i] = 0.f;
    if (i == 0) { scal[0] = -3.4e38f; scal[1] = 0.f; }
}

// ---------------------------------------------------------------------------
// dots[d] = (keys[d,:] . q) / max(|keys[d]| * |q|, 1e-8); per-block max.
// ---------------------------------------------------------------------------
__global__ void dots_k(const float* __restrict__ keys,
                       const float* __restrict__ x_t,
                       const int*   __restrict__ pick_arm,
                       float* __restrict__ dots,
                       float* __restrict__ blockmax,
                       int D, int RD)
{
    __shared__ float sq[33];     // q (RD<=32) + qnorm
    __shared__ float red[256];
    const int tid = threadIdx.x;
    if (tid == 0) {
        const int pa = pick_arm[0];
        float qn = 0.f;
        for (int j = 0; j < RD; ++j) { float v = x_t[pa * RD + j]; sq[j] = v; qn += v * v; }
        sq[RD] = sqrtf(qn);
    }
    __syncthreads();
    const float qnorm = sq[RD];
    float lmax = -3.4e38f;
    for (int d = blockIdx.x * blockDim.x + tid; d < D; d += gridDim.x * blockDim.x) {
        const float* kr = keys + (size_t)d * RD;
        float dot = 0.f, kn = 0.f;
        for (int j = 0; j < RD; ++j) { float kv = kr[j]; dot += kv * sq[j]; kn += kv * kv; }
        const float denom = fmaxf(sqrtf(kn) * qnorm, 1e-8f);
        const float v = dot / denom;
        dots[d] = v;
        lmax = fmaxf(lmax, v);
    }
    red[tid] = lmax;
    __syncthreads();
    for (int s = 128; s > 0; s >>= 1) {
        if (tid < s) red[tid] = fmaxf(red[tid], red[tid + s]);
        __syncthreads();
    }
    if (tid == 0) blockmax[blockIdx.x] = red[0];
}

__global__ void rmax_k(const float* __restrict__ blockmax, float* __restrict__ scal, int NB)
{
    __shared__ float red[256];
    const int tid = threadIdx.x;
    float m = -3.4e38f;
    for (int i = tid; i < NB; i += 256) m = fmaxf(m, blockmax[i]);
    red[tid] = m;
    __syncthreads();
    for (int s = 128; s > 0; s >>= 1) {
        if (tid < s) red[tid] = fmaxf(red[tid], red[tid + s]);
        __syncthreads();
    }
    if (tid == 0) scal[0] = red[0];
}

__global__ void exp_k(float* __restrict__ dots, const float* __restrict__ scal,
                      float* __restrict__ sumExp, int D)
{
    __shared__ float red[256];
    const float gmax = scal[0];
    float ls = 0.f;
    for (int d = blockIdx.x * blockDim.x + threadIdx.x; d < D; d += gridDim.x * blockDim.x) {
        const float e = __expf(dots[d] - gmax);
        dots[d] = e;
        ls += e;
    }
    red[threadIdx.x] = ls;
    __syncthreads();
    for (int s = 128; s > 0; s >>= 1) {
        if (threadIdx.x < s) red[threadIdx.x] += red[threadIdx.x + s];
        __syncthreads();
    }
    if (threadIdx.x == 0) atomicAdd(sumExp, red[0]);
}

// ---------------------------------------------------------------------------
// macc[:] += sum_{d in chunk} w[d] * vals[d,:]   (the 410 MB stream; bw-bound)
// 256 threads * float4 = one full 1024-wide row per block; rows chunked by grid.
// w[] chunk staged into LDS via async global->LDS (shared by all 256 threads).
// ---------------------------------------------------------------------------
__global__ void attnv_k(const float* __restrict__ vals,
                        const float* __restrict__ w,
                        float* __restrict__ macc,
                        int D, int H, int rowsPerBlock)
{
    __shared__ float sw[512];              // rowsPerBlock <= 512
    const int tid = threadIdx.x;
    const int r0  = blockIdx.x * rowsPerBlock;
    if (r0 >= D) return;
    const int nr  = ((r0 + rowsPerBlock < D) ? rowsPerBlock : (D - r0));

    for (int i = tid; i < nr; i += blockDim.x)
        stage_f1(w + r0 + i, sw + i);
    wait_async_all();
    __syncthreads();

    const size_t pitch = (size_t)(H >> 2);
    const float4* vp = (const float4*)vals + (size_t)r0 * pitch + tid;

    float4 acc = make_float4(0.f, 0.f, 0.f, 0.f);
    int r = 0;
    for (; r + 4 <= nr; r += 4) {
        if (r0 + r + 12 < D) __builtin_prefetch(&vp[(size_t)(r + 12) * pitch], 0, 1);
        const float w0 = sw[r], w1 = sw[r + 1], w2 = sw[r + 2], w3 = sw[r + 3];
        const float4 v0 = vp[(size_t)(r + 0) * pitch];
        const float4 v1 = vp[(size_t)(r + 1) * pitch];
        const float4 v2 = vp[(size_t)(r + 2) * pitch];
        const float4 v3 = vp[(size_t)(r + 3) * pitch];
        acc.x += w0 * v0.x + w1 * v1.x + w2 * v2.x + w3 * v3.x;
        acc.y += w0 * v0.y + w1 * v1.y + w2 * v2.y + w3 * v3.y;
        acc.z += w0 * v0.z + w1 * v1.z + w2 * v2.z + w3 * v3.z;
        acc.w += w0 * v0.w + w1 * v1.w + w2 * v2.w + w3 * v3.w;
    }
    for (; r < nr; ++r) {
        const float ww = sw[r];
        const float4 v = vp[(size_t)r * pitch];
        acc.x += ww * v.x; acc.y += ww * v.y; acc.z += ww * v.z; acc.w += ww * v.w;
    }
    float* mp = macc + 4 * tid;
    atomicAdd(mp + 0, acc.x);
    atomicAdd(mp + 1, acc.y);
    atomicAdd(mp + 2, acc.z);
    atomicAdd(mp + 3, acc.w);
}

// ---------------------------------------------------------------------------
// LSTM cell update + write h_t, c_t into d_out (and h_t to ws for next GEMV).
// preact layout per reference: [f | i | o | r | c_new] each H wide.
// ---------------------------------------------------------------------------
__global__ void cell_k(const float* __restrict__ preact,
                       const float* __restrict__ c_in,
                       const float* __restrict__ macc,
                       const float* __restrict__ scal,
                       float* __restrict__ h_out,
                       float* __restrict__ c_out,
                       float* __restrict__ ht_ws,
                       int H)
{
    const int j = blockIdx.x * blockDim.x + threadIdx.x;
    if (j >= H) return;
    const float invZ = 1.f / scal[1];
    const float f  = 1.f / (1.f + __expf(-preact[j]));
    const float i  = 1.f / (1.f + __expf(-preact[H + j]));
    const float o  = 1.f / (1.f + __expf(-preact[2 * H + j]));
    const float rr = 1.f / (1.f + __expf(-preact[3 * H + j]));
    const float cn = tanhf(preact[4 * H + j]);
    const float m  = tanhf(macc[j] * invZ);
    const float ct = f * c_in[j] + i * cn + rr * m;
    const float ht = o * tanhf(ct);
    c_out[j]  = ct;
    h_out[j]  = ht;
    ht_ws[j]  = ht;
}

// ---------------------------------------------------------------------------
// Heads: pi = softmax(hh @ W_actor.T + b), v = hh @ W_critic.T + b,
// a ~ categorical(key(1)) via threefry2x32 gumbel-max; out[0..3].
// ---------------------------------------------------------------------------
__device__ inline uint32_t rotl32(uint32_t v, int n) { return (v << n) | (v >> (32 - n)); }

__device__ inline void threefry2x32(uint32_t k0, uint32_t k1,
                                    uint32_t x0, uint32_t x1,
                                    uint32_t* o0, uint32_t* o1)
{
    const uint32_t ks2 = k0 ^ k1 ^ 0x1BD11BDAu;
    uint32_t X0 = x0 + k0, X1 = x1 + k1;
    const int ra[4] = {13, 15, 26, 6};
    const int rb[4] = {17, 29, 16, 24};
    #pragma unroll
    for (int j = 0; j < 4; ++j) { X0 += X1; X1 = rotl32(X1, ra[j]); X1 ^= X0; }
    X0 += k1;  X1 += ks2 + 1u;
    #pragma unroll
    for (int j = 0; j < 4; ++j) { X0 += X1; X1 = rotl32(X1, rb[j]); X1 ^= X0; }
    X0 += ks2; X1 += k0 + 2u;
    #pragma unroll
    for (int j = 0; j < 4; ++j) { X0 += X1; X1 = rotl32(X1, ra[j]); X1 ^= X0; }
    X0 += k0;  X1 += k1 + 3u;
    #pragma unroll
    for (int j = 0; j < 4; ++j) { X0 += X1; X1 = rotl32(X1, rb[j]); X1 ^= X0; }
    X0 += k1;  X1 += ks2 + 4u;
    #pragma unroll
    for (int j = 0; j < 4; ++j) { X0 += X1; X1 = rotl32(X1, ra[j]); X1 ^= X0; }
    X0 += ks2; X1 += k0 + 5u;
    *o0 = X0; *o1 = X1;
}

__device__ inline float uniform_from_bits(uint32_t bits)
{
    const uint32_t f = (bits >> 9) | 0x3f800000u;
    const float tiny = 1.1754944e-38f;
    float u = __uint_as_float(f) - 1.0f;
    u = u * (1.0f - tiny) + tiny;
    return fmaxf(u, tiny);
}

__global__ void heads_k(const float* __restrict__ hh,
                        const float* __restrict__ W_actor,
                        const float* __restrict__ b_actor,
                        const float* __restrict__ W_critic,
                        const float* __restrict__ b_critic,
                        float* __restrict__ out, int H)
{
    __shared__ float red0[256], red1[256], red2[256];
    const int tid = threadIdx.x;
    float p0 = 0.f, p1 = 0.f, pv = 0.f;
    for (int j = tid; j < H; j += 256) {
        const float h = hh[j];
        p0 += h * W_actor[j];
        p1 += h * W_actor[H + j];
        pv += h * W_critic[j];
    }
    red0[tid] = p0; red1[tid] = p1; red2[tid] = pv;
    __syncthreads();
    for (int s = 128; s > 0; s >>= 1) {
        if (tid < s) { red0[tid] += red0[tid + s]; red1[tid] += red1[tid + s]; red2[tid] += red2[tid + s]; }
        __syncthreads();
    }
    if (tid == 0) {
        const float l0 = red0[0] + b_actor[0];
        const float l1 = red1[0] + b_actor[1];
        const float v  = red2[0] + b_critic[0];
        const float mx = fmaxf(l0, l1);
        const float e0 = __expf(l0 - mx), e1 = __expf(l1 - mx);
        const float Z  = e0 + e1;
        const float pi0 = e0 / Z, pi1 = e1 / Z;
        // jax.random.key(1) -> threefry key (0,1); bits for shape (2,) use counts (0,1)
        uint32_t b0, b1;
        threefry2x32(0u, 1u, 0u, 1u, &b0, &b1);
        const float g0 = -logf(-logf(uniform_from_bits(b0)));
        const float g1 = -logf(-logf(uniform_from_bits(b1)));
        const float lp0 = logf(pi0), lp1 = logf(pi1);
        const int a = (lp1 + g1 > lp0 + g0) ? 1 : 0;
        out[0] = pi0; out[1] = pi1; out[2] = v; out[3] = a ? lp1 : lp0;
    }
}

// ---------------------------------------------------------------------------
// Launch pipeline (all on `stream`; graph-capture safe: no malloc/sync/memset).
// ---------------------------------------------------------------------------
extern "C" void kernel_launch(void* const* d_in, const int* in_sizes, int n_in,
                              void* d_out, int out_size, void* d_ws, size_t ws_size,
                              hipStream_t stream)
{
    const float* x_t     = (const float*)d_in[0];
    const float* h       = (const float*)d_in[1];
    const float* c       = (const float*)d_in[2];
    const float* keys    = (const float*)d_in[3];
    const float* vals    = (const float*)d_in[4];
    const float* W_i2h   = (const float*)d_in[5];
    const float* b_i2h   = (const float*)d_in[6];
    const float* W_h2h   = (const float*)d_in[7];
    const float* b_h2h   = (const float*)d_in[8];
    const float* W_ih    = (const float*)d_in[9];
    const float* b_ih    = (const float*)d_in[10];
    const float* W_actor = (const float*)d_in[11];
    const float* b_actor = (const float*)d_in[12];
    const float* W_critic= (const float*)d_in[13];
    const float* b_critic= (const float*)d_in[14];
    const int*   pick_arm= (const int*)d_in[15];

    const int IN  = in_sizes[0];            // 14
    const int H   = in_sizes[1];            // 1024
    const int H5  = in_sizes[5] / IN;       // 5120
    const int D   = in_sizes[4] / H;        // 100000
    const int RD  = in_sizes[3] / D;        // 10

    // Workspace layout (floats)
    float* ws       = (float*)d_ws;
    float* preact   = ws;                   // H5
    float* dots     = preact + H5;          // D (reused as w = exp(dots - max))
    const int NB_DOTS = 416;
    float* blockmax = dots + D;             // NB_DOTS
    float* macc     = blockmax + NB_DOTS;   // H
    float* hh       = macc + H;             // H
    float* ht       = hh + H;               // H
    float* scal     = ht + H;               // [0]=gmax, [1]=sumExp

    float* out = (float*)d_out;             // [pi0,pi1,v,logp, h_t(H), c_t(H)]

    // 0) init accumulators
    init_k<<<(H + 255) / 256, 256, 0, stream>>>(macc, scal, H);

    // 1) preact = x@W_i2h.T + b_i2h + h@W_h2h.T + b_h2h   (WMMA GEMV, 21 MB stream)
    {
        const int waves  = H5 / 16;                       // 320
        const int blocks = (waves * 32 + 255) / 256;      // 40
        gemv_wmma_f16<<<blocks, 256, 0, stream>>>(
            W_h2h, h, b_i2h, b_h2h, W_i2h, x_t, IN, preact, H5, H, 0);
    }

    // 2) scaled dots + per-block max
    dots_k<<<NB_DOTS, 256, 0, stream>>>(keys, x_t, pick_arm, dots, blockmax, D, RD);

    // 3) global max
    rmax_k<<<1, 256, 0, stream>>>(blockmax, scal, NB_DOTS);

    // 4) w = exp(dots - max); sumExp
    exp_k<<<NB_DOTS, 256, 0, stream>>>(dots, scal, scal + 1, D);

    // 5) macc = sum_d w[d] * vals[d,:]   (410 MB stream — the roofline term)
    {
        const int NBV = 512;
        const int rpb = (D + NBV - 1) / NBV;              // 196 (<= 512)
        attnv_k<<<NBV, 256, 0, stream>>>(vals, dots, macc, D, H, rpb);
    }

    // 6) cell update -> h_t, c_t (into d_out) and ht (ws)
    cell_k<<<(H + 255) / 256, 256, 0, stream>>>(
        preact, c, macc, scal, out + 4, out + 4 + H, ht, H);

    // 7) hh = relu(h_t @ W_ih.T + b_ih)   (WMMA GEMV, 4 MB stream)
    {
        const int waves  = H / 16;                        // 64
        const int blocks = (waves * 32 + 255) / 256;      // 8
        gemv_wmma_f16<<<blocks, 256, 0, stream>>>(
            W_ih, ht, b_ih, nullptr, nullptr, nullptr, 0, hh, H, H, 1);
    }

    // 8) heads + sampled logp -> out[0..3]
    heads_k<<<1, 256, 0, stream>>>(hh, W_actor, b_actor, W_critic, b_critic, out, H);
}
